// nonlocal_attention_7516192768081
// MI455X (gfx1250) — compile-verified, matrix operand order checked by reading, not run on hardware
//
#include <hip/hip_runtime.h>
#include <math.h>

typedef __attribute__((ext_vector_type(16))) _Float16 v16h;
typedef __attribute__((ext_vector_type(8)))  _Float16 v8h;
typedef __attribute__((ext_vector_type(8)))  float    v8f;

#define Cc   60
#define CI   30
#define NN   2304   // 48*48
#define NGRP 30

// ---------------------------------------------------------------------------
// Kernel 1: per-(b,t) image GroupNorm (30 groups, eps=1e-6, affine) + the
// three 1x1 convs (g / theta / phi), emitting f16 operands laid out for WMMA:
//   theta_ws : [6][2304][32]  (channel-contiguous per pixel, ch 30,31 = 0)
//   phi_ws   : [6][2304][32]  (same layout)
//   g_ws     : [6][32][2304]  (channel-major, rows 30,31 = 0)
// ---------------------------------------------------------------------------
__global__ __launch_bounds__(256) void gn_proj_kernel(
    const float* __restrict__ x,
    const float* __restrict__ gamma, const float* __restrict__ beta,
    const float* __restrict__ g_w,  const float* __restrict__ g_b,
    const float* __restrict__ th_w, const float* __restrict__ th_b,
    const float* __restrict__ ph_w, const float* __restrict__ ph_b,
    _Float16* __restrict__ theta_ws,
    _Float16* __restrict__ g_ws,
    _Float16* __restrict__ phi_ws)
{
  __shared__ float gwS[CI*Cc], twS[CI*Cc], pwS[CI*Cc];
  __shared__ float gbS[CI], tbS[CI], pbS[CI];
  __shared__ float gammaS[Cc], betaS[Cc];
  __shared__ float muS[NGRP], rsS[NGRP];
  __shared__ float red[256], red2[256];

  const int bt  = blockIdx.x;      // 0..5 = b*3 + t
  const int tid = threadIdx.x;

  for (int i = tid; i < CI*Cc; i += 256) { gwS[i] = g_w[i]; twS[i] = th_w[i]; pwS[i] = ph_w[i]; }
  if (tid < CI) { gbS[tid] = g_b[tid]; tbS[tid] = th_b[tid]; pbS[tid] = ph_b[tid]; }
  if (tid < Cc) { gammaS[tid] = gamma[tid]; betaS[tid] = beta[tid]; }

  const float* xi = x + (size_t)bt * Cc * NN;

  for (int grp = 0; grp < NGRP; ++grp) {
    const float* xg = xi + (size_t)grp * 2 * NN;
    float s = 0.f, s2 = 0.f;
    for (int i = tid; i < 2 * NN; i += 256) { float v = xg[i]; s += v; s2 += v * v; }
    red[tid] = s; red2[tid] = s2; __syncthreads();
    for (int off = 128; off > 0; off >>= 1) {
      if (tid < off) { red[tid] += red[tid + off]; red2[tid] += red2[tid + off]; }
      __syncthreads();
    }
    if (tid == 0) {
      float mu  = red[0] * (1.f / (2 * NN));
      float var = red2[0] * (1.f / (2 * NN)) - mu * mu;
      muS[grp] = mu; rsS[grp] = rsqrtf(var + 1e-6f);
    }
    __syncthreads();
  }

  for (int n = tid; n < NN; n += 256) {
    float hn[Cc];
#pragma unroll
    for (int c = 0; c < Cc; ++c) {
      float v  = xi[(size_t)c * NN + n];
      int grp  = c >> 1;
      hn[c] = (v - muS[grp]) * rsS[grp] * gammaS[c] + betaS[c];
    }
    _Float16* to = theta_ws + ((size_t)bt * NN + n) * 32;
    _Float16* po = phi_ws   + ((size_t)bt * NN + n) * 32;
    for (int o = 0; o < CI; ++o) {
      float at = tbS[o], ag = gbS[o], ap = pbS[o];
      const float* twr = &twS[o * Cc];
      const float* gwr = &gwS[o * Cc];
      const float* pwr = &pwS[o * Cc];
#pragma unroll
      for (int c = 0; c < Cc; ++c) {
        at += twr[c] * hn[c];
        ag += gwr[c] * hn[c];
        ap += pwr[c] * hn[c];
      }
      to[o] = (_Float16)at;
      po[o] = (_Float16)ap;
      g_ws[((size_t)bt * 32 + o) * NN + n] = (_Float16)ag;
    }
    to[30] = (_Float16)0.f; to[31] = (_Float16)0.f;
    po[30] = (_Float16)0.f; po[31] = (_Float16)0.f;
    g_ws[((size_t)bt * 32 + 30) * NN + n] = (_Float16)0.f;
    g_ws[((size_t)bt * 32 + 31) * NN + n] = (_Float16)0.f;
  }
}

// ---------------------------------------------------------------------------
// Kernel 2: fused non-local attention, transposed formulation, two-pass
// softmax over LDS-resident operands.
//   S^T = A_theta x B_phi  -> lane owns one row r, VGPRs hold m values.
//   Pass 1: S-WMMAs + register max3 chains -> true row max (one shfl_xor
//           in the whole kernel to merge the two half-wave m-subsets).
//   Pass 2: p = exp(s - mx) needs no rescaling; P (f16) in S^T D-layout IS
//           the B operand of  y^T = A_g x B_P  directly; y^T D-layout keeps
//           lane == row so normalization is a per-lane scalar.
// theta (144KB) + g (144KB) + phi rows live in LDS, filled with
// global_load_async_to_lds_b128 (ASYNCcnt path).
// y_ws: [12][2304][32] f32, img = (q*2+b)*3+t.
// ---------------------------------------------------------------------------
__global__ __launch_bounds__(256) void attn_kernel(
    const _Float16* __restrict__ theta_ws,
    const _Float16* __restrict__ g_ws,
    const _Float16* __restrict__ phi_ws,
    const float* __restrict__ stdp,
    float* __restrict__ y_ws)
{
  extern __shared__ char smem[];
  _Float16* thetaS = (_Float16*)smem;         // [2304][32]   LDS byte offset 0
  _Float16* gS     = thetaS + NN * 32;        // [32][2304]   offset 147456
  _Float16* phiS   = gS + 32 * NN;            // [128][32]    offset 294912

  const int tid   = threadIdx.x;
  const int chunk = blockIdx.x % 18;
  const int img   = blockIdx.x / 18;          // 0..11
  const int q     = img / 6;
  const int b     = (img % 6) / 3;
  const int t     = img % 3;
  const int kv_img  = b * 3 + t;
  const int q_img   = b * 3 + q;              // i==1: q indexes x[:,0], x[:,1]
  const int out_img = (q * 2 + b) * 3 + t;

  // ---- async LDS fill (dynamic LDS base is 0: no static LDS here)
  {
    const char* gsrcT = (const char*)(theta_ws + (size_t)kv_img * NN * 32);
    const char* gsrcG = (const char*)(g_ws + (size_t)kv_img * 32 * NN);
    const char* gsrcP = (const char*)(phi_ws + ((size_t)q_img * NN + chunk * 128) * 32);
    for (int i = tid; i < NN * 32 / 8; i += 256) {      // theta: 9216 x 16B
      unsigned l = (unsigned)i * 16u;
      const char* g = gsrcT + (size_t)i * 16u;
      asm volatile("global_load_async_to_lds_b128 %0, %1, off" :: "v"(l), "v"(g) : "memory");
    }
    for (int i = tid; i < NN * 32 / 8; i += 256) {      // g: 9216 x 16B
      unsigned l = 147456u + (unsigned)i * 16u;
      const char* g = gsrcG + (size_t)i * 16u;
      asm volatile("global_load_async_to_lds_b128 %0, %1, off" :: "v"(l), "v"(g) : "memory");
    }
    for (int i = tid; i < 128 * 32 / 8; i += 256) {     // phi rows: 512 x 16B
      unsigned l = 294912u + (unsigned)i * 16u;
      const char* g = gsrcP + (size_t)i * 16u;
      asm volatile("global_load_async_to_lds_b128 %0, %1, off" :: "v"(l), "v"(g) : "memory");
    }
    asm volatile("s_wait_asynccnt 0x0" ::: "memory");
  }
  __syncthreads();

  const int wave = tid >> 5;
  const int lane = tid & 31;
  const int hi   = lane >> 4;     // half-wave: selects the K-subset this lane feeds
  const int ln   = lane & 15;
  const int rowbase = chunk * 128 + wave * 16;

  union V16 { v16h v; v8h h[2]; };

  // B operand: phi for this wave's 16 rows (loop invariant). Lane -> col r.
  V16 Bphi;
  {
    const _Float16* pp = phiS + (size_t)(wave * 16 + ln) * 32;
    Bphi.h[0] = *(const v8h*)(pp + hi * 8);
    Bphi.h[1] = *(const v8h*)(pp + 16 + hi * 8);
  }

  const int   r   = rowbase + ln;           // this lane's attention row
  const float rxf = (float)(r % 48);
  const float ryf = (float)(r / 48);

  const float sv     = stdp[0];
  const float inv2s2 = 0.5f / (sv * sv);

  // =========================== pass 1: row max ===========================
  float mx = -3.0e38f;
  for (int m0 = 0; m0 < NN; m0 += 64) {
    V16 At;
    v8f z = {};
#pragma unroll
    for (int sub = 0; sub < 4; ++sub) {
      const _Float16* tp = thetaS + (size_t)(m0 + sub * 16 + ln) * 32;
      At.h[0] = *(const v8h*)(tp + hi * 8);
      At.h[1] = *(const v8h*)(tp + 16 + hi * 8);
      v8f s = __builtin_amdgcn_wmma_f32_16x16x32_f16(false, At.v, false, Bphi.v, (short)0, z, false, false);
#pragma unroll
      for (int v = 0; v < 8; ++v) mx = fmaxf(mx, s[v]);
    }
  }
  mx = fmaxf(mx, __shfl_xor(mx, 16, 32));   // the only cross-lane op before the end

  // ====================== pass 2: exp + accumulate =======================
  float lr = 0.f;
  v8f yT0 = {}; v8f yT1 = {};

  for (int m0 = 0; m0 < NN; m0 += 32) {
    // S^T tiles: lane holds row r, VGPR v -> m = m0(+16) + v + 8*hi
    v8f s0, s1;
    {
      V16 At;
      const _Float16* tp = thetaS + (size_t)(m0 + ln) * 32;
      At.h[0] = *(const v8h*)(tp + hi * 8);
      At.h[1] = *(const v8h*)(tp + 16 + hi * 8);
      v8f z = {};
      s0 = __builtin_amdgcn_wmma_f32_16x16x32_f16(false, At.v, false, Bphi.v, (short)0, z, false, false);
      const _Float16* tp2 = thetaS + (size_t)(m0 + 16 + ln) * 32;
      At.h[0] = *(const v8h*)(tp2 + hi * 8);
      At.h[1] = *(const v8h*)(tp2 + 16 + hi * 8);
      v8f z2 = {};
      s1 = __builtin_amdgcn_wmma_f32_16x16x32_f16(false, At.v, false, Bphi.v, (short)0, z2, false, false);
    }

    // issue g-tile LDS loads early so they hide under the exp work
    V16 G0, G1;
    {
      const _Float16* gp0 = gS + (size_t)ln * NN + m0;         // channels 0..15
      G0.h[0] = *(const v8h*)(gp0 + hi * 8);
      G0.h[1] = *(const v8h*)(gp0 + 16 + hi * 8);
      const _Float16* gp1 = gS + (size_t)(16 + ln) * NN + m0;  // channels 16..31 (30,31 zero)
      G1.h[0] = *(const v8h*)(gp1 + hi * 8);
      G1.h[1] = *(const v8h*)(gp1 + 16 + hi * 8);
    }

    // exp + distance weight; pack P directly into the y^T B operand
    V16 Pv;
    float psum = 0.f;
    {
      int   im = m0 + 8 * hi;
      float xm = (float)(im % 48), ym = (float)(im / 48);
#pragma unroll
      for (int v = 0; v < 8; ++v) {
        float p = __expf(s0[v] - mx);
        psum += p;
        float dx = rxf - xm, dy = ryf - ym;
        float w = __expf(-inv2s2 * (dx * dx + dy * dy));
        Pv.h[0][v] = (_Float16)(p * w);
        float nx = xm + 1.f;
        int wrap = (nx > 47.5f);
        xm = wrap ? 0.f : nx;
        ym = wrap ? ym + 1.f : ym;
      }
      im = m0 + 16 + 8 * hi;
      xm = (float)(im % 48); ym = (float)(im / 48);
#pragma unroll
      for (int v = 0; v < 8; ++v) {
        float p = __expf(s1[v] - mx);
        psum += p;
        float dx = rxf - xm, dy = ryf - ym;
        float w = __expf(-inv2s2 * (dx * dx + dy * dy));
        Pv.h[1][v] = (_Float16)(p * w);
        float nx = xm + 1.f;
        int wrap = (nx > 47.5f);
        xm = wrap ? 0.f : nx;
        ym = wrap ? ym + 1.f : ym;
      }
    }
    lr += psum;   // this lane's disjoint m-subset; merged once at the end

    // y^T += g^T x P (A = g channel-major tiles, B = P already in layout)
    yT0 = __builtin_amdgcn_wmma_f32_16x16x32_f16(false, G0.v, false, Pv.v, (short)0, yT0, false, false);
    yT1 = __builtin_amdgcn_wmma_f32_16x16x32_f16(false, G1.v, false, Pv.v, (short)0, yT1, false, false);
  }

  // merge denominators of the two half-wave m-subsets, normalize, store.
  lr += __shfl_xor(lr, 16, 32);
  float invl = 1.f / lr;
  float* yo = y_ws + ((size_t)out_img * NN + r) * 32 + 8 * hi;   // ch = 8*hi + v
  float4 f0 = make_float4(yT0[0] * invl, yT0[1] * invl, yT0[2] * invl, yT0[3] * invl);
  float4 f1 = make_float4(yT0[4] * invl, yT0[5] * invl, yT0[6] * invl, yT0[7] * invl);
  float4 f2 = make_float4(yT1[0] * invl, yT1[1] * invl, yT1[2] * invl, yT1[3] * invl);
  float4 f3 = make_float4(yT1[4] * invl, yT1[5] * invl, yT1[6] * invl, yT1[7] * invl);
  *(float4*)(yo)      = f0;
  *(float4*)(yo + 4)  = f1;
  *(float4*)(yo + 16) = f2;   // ch = 16 + 8*hi + v
  *(float4*)(yo + 20) = f3;
}

// ---------------------------------------------------------------------------
// Kernel 3: z[b,q,c,n] = wz_w[c, t*30+ci] . y[q,b,t,n,ci] + wz_b[c] + x[b,q,c,n]
// ---------------------------------------------------------------------------
__global__ __launch_bounds__(256) void zout_kernel(
    const float* __restrict__ x,
    const float* __restrict__ wz_w, const float* __restrict__ wz_b,
    const float* __restrict__ y_ws,
    float* __restrict__ out)
{
  __shared__ float wzS[Cc * 90];
  __shared__ float wbS[Cc];
  const int tid = threadIdx.x;
  for (int i = tid; i < Cc * 90; i += 256) wzS[i] = wz_w[i];
  if (tid < Cc) wbS[tid] = wz_b[tid];
  __syncthreads();

  int idx = blockIdx.x * 256 + tid;
  if (idx >= 2 * 2 * NN * Cc) return;
  int c  = idx % Cc;
  int n  = (idx / Cc) % NN;
  int qb = idx / (Cc * NN);
  int q  = qb & 1;
  int b  = qb >> 1;

  float acc = wbS[c];
  const float* wr = &wzS[c * 90];
#pragma unroll
  for (int t = 0; t < 3; ++t) {
    const float* yb = y_ws + (((size_t)((q * 2 + b) * 3 + t)) * NN + n) * 32;
    for (int ci = 0; ci < CI; ++ci) acc += wr[t * 30 + ci] * yb[ci];
  }
  out[((size_t)(b * 2 + q) * Cc + c) * NN + n] =
      acc + x[((size_t)(b * 3 + q) * Cc + c) * NN + n];
}

// ---------------------------------------------------------------------------
extern "C" void kernel_launch(void* const* d_in, const int* in_sizes, int n_in,
                              void* d_out, int out_size, void* d_ws, size_t ws_size,
                              hipStream_t stream)
{
  (void)in_sizes; (void)n_in; (void)out_size; (void)ws_size;
  const float* x     = (const float*)d_in[0];
  const float* gamma = (const float*)d_in[1];
  const float* beta  = (const float*)d_in[2];
  const float* g_w   = (const float*)d_in[3];
  const float* g_b   = (const float*)d_in[4];
  const float* th_w  = (const float*)d_in[5];
  const float* th_b  = (const float*)d_in[6];
  const float* ph_w  = (const float*)d_in[7];
  const float* ph_b  = (const float*)d_in[8];
  const float* wz_w  = (const float*)d_in[9];
  const float* wz_b  = (const float*)d_in[10];
  const float* stdp  = (const float*)d_in[11];
  // d_in[12] == i (setup fixes i=1; grid shape hardcoded for that branch)

  char* ws = (char*)d_ws;
  const size_t PROJ_BYTES = (size_t)6 * NN * 32 * sizeof(_Float16);   // 884736
  _Float16* theta_ws = (_Float16*)(ws);
  _Float16* g_ws     = (_Float16*)(ws + PROJ_BYTES);
  _Float16* phi_ws   = (_Float16*)(ws + 2 * PROJ_BYTES);
  float*    y_ws     = (float*)   (ws + 3 * PROJ_BYTES);              // 12*2304*32 f32

  gn_proj_kernel<<<6, 256, 0, stream>>>(x, gamma, beta, g_w, g_b, th_w, th_b,
                                        ph_w, ph_b, theta_ws, g_ws, phi_ws);

  const size_t smem = (size_t)(NN * 32 + 32 * NN + 128 * 32) * sizeof(_Float16); // 296 KB
  hipFuncSetAttribute((const void*)attn_kernel,
                      hipFuncAttributeMaxDynamicSharedMemorySize, (int)smem);
  attn_kernel<<<12 * 18, 256, smem, stream>>>(theta_ws, g_ws, phi_ws, stdp, y_ws);

  zout_kernel<<<(2 * 2 * NN * Cc) / 256, 256, 0, stream>>>(x, wz_w, wz_b, y_ws, (float*)d_out);
}
